// DynamicGraphEmbedding_10307921510690
// MI455X (gfx1250) — matrix operand order repaired and checked by reading, not compiled, not run on hardware
//
#include <hip/hip_runtime.h>
#include <hip/hip_bf16.h>
#include <stdint.h>

// Problem constants (from reference): B=8, N=2048, D=1024, H=512, k=5
#define BN 8
#define NN 2048
#define DD 1024
#define HH 512
#define KNB 5
#define CHUNK 256
#define LDPA (DD + 8)   // padded LDS leading dim (bf16 elems): 2064B stride = 4-bank rotation

typedef __attribute__((ext_vector_type(16))) __bf16 v16bf;
typedef __attribute__((ext_vector_type(8)))  __bf16 v8bf;
typedef __attribute__((ext_vector_type(4)))  __bf16 v4bf;
typedef __attribute__((ext_vector_type(8)))  float  v8f;

static __device__ __forceinline__ __bf16 f32_to_bf16(float f) {
  union { float f; uint32_t u; } v; v.f = f;
  uint32_t r = v.u + 0x7FFFu + ((v.u >> 16) & 1u);   // round-to-nearest-even
  uint16_t h = (uint16_t)(r >> 16);
  return __builtin_bit_cast(__bf16, h);
}

// ---------------- CDNA5 async global->LDS staging (ASYNCcnt path) ----------------
// Copy a 16-row x (rowBytes) strip into LDS with padded row stride ldsRowBytes.
// Each thread issues strided b128 async transfers; completion via s_wait_asynccnt.
static __device__ __forceinline__ void async_stage_strip(const char* g, int rowBytes,
                                                         char* lds, int ldsRowBytes) {
  int tid = threadIdx.x;                       // 256 threads
  int chunksPerRow = rowBytes >> 4;            // 16B chunks
  int total = 16 * chunksPerRow;
  for (int c = tid; c < total; c += 256) {
    int row = c / chunksPerRow;
    int off = (c - row * chunksPerRow) << 4;
    unsigned la = (unsigned)(uintptr_t)(lds + row * ldsRowBytes + off); // LDS byte addr
    const char* ga = g + (size_t)row * rowBytes + off;
    asm volatile("global_load_async_to_lds_b128 %0, %1, off"
                 :: "v"(la), "v"(ga) : "memory");
  }
#if __has_builtin(__builtin_amdgcn_s_wait_asynccnt)
  __builtin_amdgcn_s_wait_asynccnt(0);
#else
  asm volatile("s_wait_asynccnt 0" ::: "memory");
#endif
}

// ---- WMMA fragment loaders (16x16x32 bf16, wave32; layouts per ISA 7.12.2) ----
// A: 16(M) x 32(K), row-major source with leading dim ld (works for LDS or global).
// lane m (0-15): K = k..k+7 and k+16..k+23 ; lane m+16: K = k+8..k+15 and k+24..k+31
static __device__ __forceinline__ v16bf load_frag_a(const __bf16* src, int ld, int k) {
  int lane = threadIdx.x & 31;
  int row  = lane & 15;
  int kb   = k + ((lane >> 4) << 3);        // +0 or +8
  const __bf16* p = src + (size_t)row * ld + kb;
  v8bf lo = *(const v8bf*)(p);              // K = kb .. kb+7
  v8bf hi = *(const v8bf*)(p + 16);         // K = kb+16 .. kb+23
  v16bf r;
#pragma unroll
  for (int i = 0; i < 8; ++i) { r[i] = lo[i]; r[i + 8] = hi[i]; }
  return r;
}

// B: 32(K) x 16(N); column n of B is row n of src (row-major, leading dim ld).
// lane n (0-15): K = k..k+7 and k+16..k+23 ; lane n+16: K = k+8..k+15 and k+24..k+31
static __device__ __forceinline__ v16bf load_frag_b(const __bf16* __restrict__ src,
                                                    int ld, int k) {
  int lane = threadIdx.x & 31;
  int col  = lane & 15;
  int kb   = k + ((lane >> 4) << 3);        // +0 or +8
  const __bf16* p = src + (size_t)col * ld + kb;
  v8bf lo = *(const v8bf*)(p);
  v8bf hi = *(const v8bf*)(p + 16);
  v16bf r;
#pragma unroll
  for (int i = 0; i < 8; ++i) { r[i] = lo[i]; r[i + 8] = hi[i]; }
  return r;
}

static __device__ __forceinline__ v8f wmma_bf16(v16bf a, v16bf b, v8f c) {
  return __builtin_amdgcn_wmma_f32_16x16x32_bf16(false, a, false, b,
                                                 (short)0, c, false, false);
}

// ---------------- Kernel 1: row L2-normalize, f32 -> bf16 ----------------
__global__ __launch_bounds__(256) void k_rownorm(const float* __restrict__ x,
                                                 __bf16* __restrict__ xn) {
  int row = blockIdx.x;                  // 0 .. B*N-1
  int tid = threadIdx.x;                 // 256 threads, 4 elems each (D=1024)
  const float* xr = x + (size_t)row * DD;
  float4 f = ((const float4*)xr)[tid];
  float ss = f.x * f.x + f.y * f.y + f.z * f.z + f.w * f.w;
  __shared__ float red[256];
  red[tid] = ss;
  __syncthreads();
  for (int s = 128; s > 0; s >>= 1) {
    if (tid < s) red[tid] += red[tid + s];
    __syncthreads();
  }
  float inv = 1.0f / (sqrtf(red[0]) + 1e-8f);
  v4bf o;
  o[0] = f32_to_bf16(f.x * inv);
  o[1] = f32_to_bf16(f.y * inv);
  o[2] = f32_to_bf16(f.z * inv);
  o[3] = f32_to_bf16(f.w * inv);
  ((v4bf*)(xn + (size_t)row * DD))[tid] = o;
}

// ------- Kernel 2: transpose+convert weight f32[K,N] -> bf16 wT[N,K] -------
__global__ __launch_bounds__(256) void k_convT(const float* __restrict__ w,
                                               __bf16* __restrict__ wT,
                                               int K, int Nc) {
  int i = blockIdx.x * 256 + threadIdx.x;
  if (i >= K * Nc) return;
  int n = i / K;
  int k = i - n * K;
  wT[i] = f32_to_bf16(w[(size_t)k * Nc + n]);
}

// --- Kernel 3: cosine-sim GEMM (bf16 WMMA) + streaming top-5 + softmax + agg ---
// One workgroup per (batch, 16-row strip). 8 waves.
// LDS phase overlay (49408 B total, barriers separate phases):
//   loop phase : [0, 33024)  A-strip (16 x LDPA bf16, async-staged)
//                [33024, 49408) S slab (16 x CHUNK f32)
//   merge phase: [0, 10240) candV ; [10240, 20480) candI
//                [20480, 20800) wgt ; [20800, 21120) nbr (live into aggregate)
__global__ __launch_bounds__(256) void k_simtopk(const float* __restrict__ x,
                                                 const __bf16* __restrict__ xn,
                                                 __bf16* __restrict__ h) {
  __shared__ __align__(16) char smem[16 * LDPA * 2 + 16 * CHUNK * 4];
  __bf16* sA    = (__bf16*)smem;
  float*  S     = (float*)(smem + 16 * LDPA * 2);
  float*  candV = (float*)smem;                       // [16][32][KNB]
  int*    candI = (int*)(smem + 10240);               // [16][32][KNB]
  float*  wgt   = (float*)(smem + 20480);             // [16][KNB]
  int*    nbr   = (int*)(smem + 20800);               // [16][KNB]

  int b       = blockIdx.y;
  int rowBase = blockIdx.x * 16;                      // in-batch row of strip
  int tid  = threadIdx.x;
  int wave = tid >> 5;
  int lane = tid & 31;

  const __bf16* xnb = xn + (size_t)b * NN * DD;

  // stage the 16-row A strip into LDS via async copies
  async_stage_strip((const char*)(xnb + (size_t)rowBase * DD), DD * 2,
                    smem, LDPA * 2);
  __syncthreads();

  // running per-lane top-5 for rows (2*wave) and (2*wave+1) of this strip
  float tv[2][KNB];
  int   ti[2][KNB];
#pragma unroll
  for (int r = 0; r < 2; ++r)
#pragma unroll
    for (int j = 0; j < KNB; ++j) { tv[r][j] = -3.0e38f; ti[r][j] = 0; }

  for (int chunk = 0; chunk < NN / CHUNK; ++chunk) {
    // ---- compute: each wave -> two 16x16 tiles (cols c0, c0+16) ----
    int c0 = chunk * CHUNK + wave * 32;
    const __bf16* bp0 = xnb + (size_t)c0 * DD;
    const __bf16* bp1 = xnb + (size_t)(c0 + 16) * DD;
    v8f acc0 = {};
    v8f acc1 = {};
    for (int k = 0; k < DD; k += 32) {
      v16bf a = load_frag_a(sA, LDPA, k);            // from LDS, conflict-free
      acc0 = wmma_bf16(a, load_frag_b(bp0, DD, k), acc0);
      acc1 = wmma_bf16(a, load_frag_b(bp1, DD, k), acc1);
    }
    // C layout: VGPR r, lanes 0-15 -> (M=r, N=lane); lanes 16-31 -> (M=r+8, N=lane-16)
    int n  = lane & 15;
    int mB = (lane >> 4) << 3;
#pragma unroll
    for (int r = 0; r < 8; ++r) {
      S[(mB + r) * CHUNK + wave * 32 + n]      = acc0[r];
      S[(mB + r) * CHUNK + wave * 32 + 16 + n] = acc1[r];
    }
    __syncthreads();
    // ---- scan: wave owns strip rows 2*wave, 2*wave+1 ----
#pragma unroll
    for (int r = 0; r < 2; ++r) {
      int rloc = wave * 2 + r;
      int grow = rowBase + rloc;
#pragma unroll
      for (int j = 0; j < CHUNK / 32; ++j) {
        int cl   = j * 32 + lane;
        int gcol = chunk * CHUNK + cl;
        float v  = (gcol == grow) ? -3.0e38f : S[rloc * CHUNK + cl];  // mask diag
        if (v > tv[r][KNB - 1]) {
          tv[r][KNB - 1] = v; ti[r][KNB - 1] = gcol;
#pragma unroll
          for (int s = KNB - 1; s > 0; --s) {
            if (tv[r][s] > tv[r][s - 1]) {
              float fv = tv[r][s]; tv[r][s] = tv[r][s - 1]; tv[r][s - 1] = fv;
              int   fi = ti[r][s]; ti[r][s] = ti[r][s - 1]; ti[r][s - 1] = fi;
            }
          }
        }
      }
    }
    __syncthreads();
  }

  // ---- dump per-lane candidates, merge 160 -> 5 per row, softmax ----
#pragma unroll
  for (int r = 0; r < 2; ++r) {
    int rloc = wave * 2 + r;
#pragma unroll
    for (int j = 0; j < KNB; ++j) {
      candV[(rloc * 32 + lane) * KNB + j] = tv[r][j];
      candI[(rloc * 32 + lane) * KNB + j] = ti[r][j];
    }
  }
  __syncthreads();
  if (tid < 16) {
    float bv[KNB]; int bi[KNB];
#pragma unroll
    for (int j = 0; j < KNB; ++j) { bv[j] = -3.0e38f; bi[j] = 0; }
    for (int l = 0; l < 32; ++l)
#pragma unroll
      for (int j = 0; j < KNB; ++j) {
        float v = candV[(tid * 32 + l) * KNB + j];
        int idx = candI[(tid * 32 + l) * KNB + j];
        if (v > bv[KNB - 1]) {
          bv[KNB - 1] = v; bi[KNB - 1] = idx;
#pragma unroll
          for (int s = KNB - 1; s > 0; --s)
            if (bv[s] > bv[s - 1]) {
              float fv = bv[s]; bv[s] = bv[s - 1]; bv[s - 1] = fv;
              int   fi = bi[s]; bi[s] = bi[s - 1]; bi[s - 1] = fi;
            }
        }
      }
    float m = bv[0];
    float e[KNB], sum = 0.f;
#pragma unroll
    for (int j = 0; j < KNB; ++j) { e[j] = __expf(bv[j] - m); sum += e[j]; }
    float inv = 1.0f / sum;
#pragma unroll
    for (int j = 0; j < KNB; ++j) { wgt[tid * KNB + j] = e[j] * inv; nbr[tid * KNB + j] = bi[j]; }
  }
  __syncthreads();

  // ---- aggregate: h[row] = x[row] + sum_k w_k * x[nbr_k]  (f32, out bf16) ----
  const float* xb = x + (size_t)b * NN * DD;
  __bf16* hb = h + ((size_t)b * NN + rowBase) * DD;
  int rloc = tid >> 4;                      // 16 threads per row
  int cid  = tid & 15;
  const float* xr = xb + (size_t)(rowBase + rloc) * DD;
  float w0 = wgt[rloc * KNB + 0], w1 = wgt[rloc * KNB + 1], w2 = wgt[rloc * KNB + 2],
        w3 = wgt[rloc * KNB + 3], w4 = wgt[rloc * KNB + 4];
  const float* p0 = xb + (size_t)nbr[rloc * KNB + 0] * DD;
  const float* p1 = xb + (size_t)nbr[rloc * KNB + 1] * DD;
  const float* p2 = xb + (size_t)nbr[rloc * KNB + 2] * DD;
  const float* p3 = xb + (size_t)nbr[rloc * KNB + 3] * DD;
  const float* p4 = xb + (size_t)nbr[rloc * KNB + 4] * DD;
  for (int c = cid * 4; c < DD; c += 64) {
    float4 a  = *(const float4*)(xr + c);
    float4 f0 = *(const float4*)(p0 + c);
    float4 f1 = *(const float4*)(p1 + c);
    float4 f2 = *(const float4*)(p2 + c);
    float4 f3 = *(const float4*)(p3 + c);
    float4 f4 = *(const float4*)(p4 + c);
    a.x += w0 * f0.x + w1 * f1.x + w2 * f2.x + w3 * f3.x + w4 * f4.x;
    a.y += w0 * f0.y + w1 * f1.y + w2 * f2.y + w3 * f3.y + w4 * f4.y;
    a.z += w0 * f0.z + w1 * f1.z + w2 * f2.z + w3 * f3.z + w4 * f4.z;
    a.w += w0 * f0.w + w1 * f1.w + w2 * f2.w + w3 * f3.w + w4 * f4.w;
    v4bf o;
    o[0] = f32_to_bf16(a.x); o[1] = f32_to_bf16(a.y);
    o[2] = f32_to_bf16(a.z); o[3] = f32_to_bf16(a.w);
    *(v4bf*)(hb + (size_t)rloc * DD + c) = o;
  }
}

// ------- Kernels 4/5: GEMM (bf16 WMMA, f32 acc) + bias + ReLU epilogue -------
// Block = 16-row strip (async-staged into LDS); 8 waves x 4 col-tiles = Nout 512.
template <bool BF16_OUT>
__global__ __launch_bounds__(256) void k_gemm_bias_relu(
    const __bf16* __restrict__ A,     // [M, K] row-major
    const __bf16* __restrict__ WT,    // [Nout, K] (i.e. W column-major)
    const float* __restrict__ bias,   // [Nout]
    void* __restrict__ out, int K, int Nout) {
  __shared__ __align__(16) char smemA[16 * (1024 + 8) * 2];  // max K = 1024
  int ldp = K + 8;                    // padded LDS leading dim (bank rotation)
  __bf16* sA = (__bf16*)smemA;

  int rowBase = blockIdx.x * 16;
  int wave = threadIdx.x >> 5;
  int lane = threadIdx.x & 31;

  async_stage_strip((const char*)(A + (size_t)rowBase * K), K * 2, smemA, ldp * 2);
  __syncthreads();

  v8f acc[4];
  int colBase[4];
#pragma unroll
  for (int t = 0; t < 4; ++t) {
    v8f z = {};
    acc[t] = z;
    colBase[t] = (wave * 4 + t) * 16;
  }
  for (int k = 0; k < K; k += 32) {
    v16bf a = load_frag_a(sA, ldp, k);
#pragma unroll
    for (int t = 0; t < 4; ++t) {
      v16bf bf = load_frag_b(WT + (size_t)colBase[t] * K, K, k);
      acc[t] = wmma_bf16(a, bf, acc[t]);
    }
  }
  int n  = lane & 15;
  int mB = (lane >> 4) << 3;
#pragma unroll
  for (int t = 0; t < 4; ++t) {
    float bv = bias[colBase[t] + n];
#pragma unroll
    for (int r = 0; r < 8; ++r) {
      float v = acc[t][r] + bv;
      v = v > 0.f ? v : 0.f;
      size_t off = (size_t)(rowBase + mB + r) * Nout + colBase[t] + n;
      if (BF16_OUT) ((__bf16*)out)[off] = f32_to_bf16(v);
      else          ((float*)out)[off]  = v;
    }
  }
}

extern "C" void kernel_launch(void* const* d_in, const int* in_sizes, int n_in,
                              void* d_out, int out_size, void* d_ws, size_t ws_size,
                              hipStream_t stream) {
  (void)in_sizes; (void)n_in; (void)out_size; (void)ws_size;
  const float* x  = (const float*)d_in[0];
  const float* w1 = (const float*)d_in[1];
  const float* b1 = (const float*)d_in[2];
  const float* w2 = (const float*)d_in[3];
  const float* b2 = (const float*)d_in[4];
  float* out = (float*)d_out;

  // workspace layout (~85.5 MB total)
  char* ws = (char*)d_ws;
  __bf16* xn  = (__bf16*)ws; ws += (size_t)BN * NN * DD * 2;   // normalized x, bf16
  __bf16* h1  = (__bf16*)ws; ws += (size_t)BN * NN * DD * 2;   // x + agg, bf16
  __bf16* h2  = (__bf16*)ws; ws += (size_t)BN * NN * HH * 2;   // layer-1 out, bf16
  __bf16* w1T = (__bf16*)ws; ws += (size_t)DD * HH * 2;        // w1^T, bf16
  __bf16* w2T = (__bf16*)ws;                                   // w2^T, bf16

  k_rownorm<<<BN * NN, 256, 0, stream>>>(x, xn);
  k_convT<<<(DD * HH + 255) / 256, 256, 0, stream>>>(w1, w1T, DD, HH);
  k_convT<<<(HH * HH + 255) / 256, 256, 0, stream>>>(w2, w2T, HH, HH);
  k_simtopk<<<dim3(NN / 16, BN), 256, 0, stream>>>(x, xn, h1);
  k_gemm_bias_relu<true ><<<BN * NN / 16, 256, 0, stream>>>(h1, w1T, b1, h2, DD, HH);
  k_gemm_bias_relu<false><<<BN * NN / 16, 256, 0, stream>>>(h2, w2T, b2, out, HH, HH);
}
